// TransFormerLayer_68247030334341
// MI455X (gfx1250) — matrix operand order repaired and checked by reading, not compiled, not run on hardware
//
#include <hip/hip_runtime.h>
#include <math.h>

// ---------------------------------------------------------------------------
// Transformer layer (GroupAttention + MHA + FFN) for gfx1250 / MI455X.
// All matmuls run through v_wmma_f32_16x16x32_f16 (wave32 WMMA), with the
// A-tile staged in LDS via gfx1250 async global->LDS copies (double-buffered)
// and register-blocked 32x32 per-wave output tiles (4 WMMA per K-step).
// Scratch usage: ~254 MB of d_ws (linear layout, offsets below).
// Outputs (f32, concatenated): x [B,S,D] | g_attn [B,S,S] | nb [B,S,S]
// ---------------------------------------------------------------------------

#define Bc   4
#define Sc   1024
#define Dc   1024
#define Hc   16
#define DKc  64
#define DFFc 4096
#define NEGV (-1000000000.0f)

typedef _Float16 v16h __attribute__((ext_vector_type(16)));
typedef _Float16 v8h  __attribute__((ext_vector_type(8)));
typedef float    v8f  __attribute__((ext_vector_type(8)));

#if defined(__HIP_DEVICE_COMPILE__) && defined(__gfx1250__) && \
    __has_builtin(__builtin_amdgcn_global_load_async_to_lds_b128) && \
    __has_builtin(__builtin_amdgcn_s_wait_asynccnt)
#define USE_ASYNC_LDS 1
// builtin signature (from hipcc diagnostic): param1 = AS1 (global) pointer to
// gcc-vector int32x4, param2 = AS3 (LDS) pointer to int32x4, then imm offset,
// imm cpol.
typedef int v4i_g __attribute__((vector_size(16)));
typedef __attribute__((address_space(1))) v4i_g as1_v4i;
typedef __attribute__((address_space(3))) v4i_g as3_v4i;
#endif

// ---------------------------------------------------------------------------
// Batched WMMA GEMM:  C = alpha * (A @ Bt^T) + bias (+residual), optional GELU
//   A  : f16 [M,K] row-major (row stride lda, batch stride sAb)
//   Bt : f16 [N,K] row-major = B transposed (row stride ldb, batch stride sBb)
//   C  : f32 and/or f16; element (z,r,c) at cOff(z) + r*ldc + c where
//        cOff(z) = (z/cDiv)*sCo + (z%cDiv)*sCi  (head-merged epilogue support)
// Block: 256 threads = 8 waves as 4(M) x 2(N); block tile 128x64; per-wave
// 32x32 (2x2 WMMA fragments). A block-tile [128x32] is staged in LDS with
// double buffering (async global->LDS when available), B streams from global.
// Requires M % 128 == 0, N % 64 == 0, K % 32 == 0 (true for all uses here).
// Fragment layouts used here (A and B share a contiguous K order):
//   A lane l: row = l&15, 16 contiguous K at (l>>4)*16          (2 x 16B)
//   B lane l: col = l&15, 16 contiguous K at (l>>4)*16          (1 x 32B)
//   C lane l: col = l&15, rows r + 8*(l>>4) for VGPR r = 0..7
// ---------------------------------------------------------------------------
template <bool GELU>
__global__ __launch_bounds__(256) void gemm_wmma(
    const _Float16* __restrict__ A, int lda, long sAb,
    const _Float16* __restrict__ Bt, int ldb, long sBb,
    const float* __restrict__ bias,
    const float* __restrict__ residual,
    float* __restrict__ Cf, _Float16* __restrict__ Ch,
    int ldc, int cDiv, long sCo, long sCi,
    int M, int N, int K, float alpha)
{
  __shared__ _Float16 atile[2][128 * 32];   // 2 x 8 KB double buffer

  const int z    = blockIdx.z;
  const long cOff = (long)(z / cDiv) * sCo + (long)(z % cDiv) * sCi;
  const _Float16* Az = A  + (long)z * sAb;
  const _Float16* Bz = Bt + (long)z * sBb;

  const int nBlks = N >> 6;
  const int bm = blockIdx.x / nBlks;
  const int bn = blockIdx.x - bm * nBlks;
  const int mBase = bm << 7;                 // 128-row block tile
  const int nBase = bn << 6;                 // 64-col  block tile

  const int tid   = threadIdx.x;
  const int wave  = tid >> 5;
  const int lane  = tid & 31;
  const int waveM = wave & 3;                // 0..3
  const int waveN = wave >> 2;               // 0..1
  const int half  = lane >> 4;               // 0 | 1
  const int l16   = lane & 15;

  // ---- stage one 128x32 A tile into LDS (one b128 x2 per thread) ----
  auto stage = [&](int buf, int k0) {
#pragma unroll
    for (int c = 0; c < 2; ++c) {
      const int chunk = tid * 2 + c;         // 512 x 16B chunks
      const int r = chunk >> 2;
      const int o = (chunk & 3) * 8;
      const _Float16* g = Az + (long)(mBase + r) * lda + k0 + o;
      _Float16* l = &atile[buf][r * 32 + o];
#ifdef USE_ASYNC_LDS
      __builtin_amdgcn_global_load_async_to_lds_b128(
          (as1_v4i*)(size_t)g, (as3_v4i*)(unsigned)(size_t)l, 0, 0);
#else
      *(v8h*)l = *(const v8h*)g;
#endif
    }
  };

  // per-wave B row pointers (two 16-col fragments)
  const int col0 = nBase + waveN * 32 + l16;
  const int col1 = col0 + 16;
  const _Float16* brow0 = Bz + (long)col0 * ldb + half * 16;
  const _Float16* brow1 = Bz + (long)col1 * ldb + half * 16;
  // per-wave A fragment base offsets inside the LDS tile
  const int aoff0 = (waveM * 32 + l16) * 32 + half * 16;       // mi = 0
  const int aoff1 = aoff0 + 16 * 32;                           // mi = 1

  v8f acc00 = {}, acc01 = {}, acc10 = {}, acc11 = {};

  int buf = 0;
  stage(0, 0);
  for (int k0 = 0; k0 < K; k0 += 32) {
#ifdef USE_ASYNC_LDS
    __builtin_amdgcn_s_wait_asynccnt(0);     // current tile arrived
#endif
    __syncthreads();                         // ...and prior reads finished
    if (k0 + 32 < K) stage(buf ^ 1, k0 + 32);

    __builtin_prefetch((const void*)(brow0 + k0 + 256), 0, 1);
    __builtin_prefetch((const void*)(brow1 + k0 + 256), 0, 1);

    v16h b0 = *(const v16h*)(brow0 + k0);
    v16h b1 = *(const v16h*)(brow1 + k0);

    const _Float16* at0 = &atile[buf][aoff0];
    const _Float16* at1 = &atile[buf][aoff1];
    v8h a0lo = *(const v8h*)(at0);
    v8h a0hi = *(const v8h*)(at0 + 8);
    v8h a1lo = *(const v8h*)(at1);
    v8h a1hi = *(const v8h*)(at1 + 8);
    v16h a0, a1;
#pragma unroll
    for (int i = 0; i < 8; ++i) {
      a0[i] = a0lo[i]; a0[i + 8] = a0hi[i];
      a1[i] = a1lo[i]; a1[i + 8] = a1hi[i];
    }
    acc00 = __builtin_amdgcn_wmma_f32_16x16x32_f16(false, a0, false, b0, (short)0, acc00, false, false);
    acc01 = __builtin_amdgcn_wmma_f32_16x16x32_f16(false, a0, false, b1, (short)0, acc01, false, false);
    acc10 = __builtin_amdgcn_wmma_f32_16x16x32_f16(false, a1, false, b0, (short)0, acc10, false, false);
    acc11 = __builtin_amdgcn_wmma_f32_16x16x32_f16(false, a1, false, b1, (short)0, acc11, false, false);
    buf ^= 1;
  }

  // ---- epilogue ----
  const float bc0 = bias ? bias[col0] : 0.0f;
  const float bc1 = bias ? bias[col1] : 0.0f;
  const v8f* accs[4] = { &acc00, &acc01, &acc10, &acc11 };
#pragma unroll
  for (int mi = 0; mi < 2; ++mi) {
#pragma unroll
    for (int ni = 0; ni < 2; ++ni) {
      const v8f& a = *accs[mi * 2 + ni];
      const int col = ni ? col1 : col0;
      const float bcol = ni ? bc1 : bc0;
#pragma unroll
      for (int r = 0; r < 8; ++r) {
        const int row = mBase + waveM * 32 + mi * 16 + half * 8 + r;
        const long idx = cOff + (long)row * ldc + col;
        float v = a[r] * alpha + bcol;
        if (residual) v += residual[idx];
        if (GELU) v = 0.5f * v * (1.0f + erff(v * 0.70710678118654752f));
        if (Cf) Cf[idx] = v;
        if (Ch) Ch[idx] = (_Float16)v;
      }
    }
  }
}

// --------------------------- small helper kernels --------------------------

__device__ __forceinline__ float blk_reduce(float v, float* sh, bool is_max) {
  const int t = threadIdx.x;
  sh[t] = v; __syncthreads();
  for (int off = 128; off > 0; off >>= 1) {
    if (t < off) sh[t] = is_max ? fmaxf(sh[t], sh[t + off]) : (sh[t] + sh[t + off]);
    __syncthreads();
  }
  const float r = sh[0];
  __syncthreads();
  return r;
}

// torch-style LayerNorm (ddof=1, std+eps); writes f16 for the WMMA GEMMs.
__global__ __launch_bounds__(256) void ln_kernel(
    const float* __restrict__ x, const float* __restrict__ a,
    const float* __restrict__ b, _Float16* __restrict__ outh, int D)
{
  __shared__ float sh[256];
  const long row = blockIdx.x;
  const float* xr = x + row * D;
  float s = 0.f, s2 = 0.f;
  for (int j = threadIdx.x; j < D; j += 256) { float v = xr[j]; s += v; s2 += v * v; }
  s  = blk_reduce(s,  sh, false);
  s2 = blk_reduce(s2, sh, false);
  const float mean = s / D;
  const float var  = (s2 - D * mean * mean) / (D - 1);
  const float inv  = 1.0f / (sqrtf(var) + 1e-6f);
  for (int j = threadIdx.x; j < D; j += 256)
    outh[row * D + j] = (_Float16)(a[j] * (xr[j] - mean) * inv + b[j]);
}

// f32 [K,N] -> f16 [N,K] (weight transpose+convert, done once)
__global__ __launch_bounds__(256) void tconv_kernel(
    const float* __restrict__ W, _Float16* __restrict__ Wt, int K, int N)
{
  long idx = (long)blockIdx.x * 256 + threadIdx.x;
  if (idx >= (long)K * N) return;
  const int k = (int)(idx / N), n = (int)(idx - (long)k * N);
  Wt[(long)n * K + k] = (_Float16)W[idx];
}

// GroupAttention band softmax (mask = eos & (|i-j|==1)), in place.
__global__ __launch_bounds__(256) void ga_softmax_kernel(
    float* __restrict__ sc, const int* __restrict__ eos)
{
  __shared__ float sh[256];
  const int i = blockIdx.x, b = blockIdx.y;
  float* row = sc + ((long)b * Sc + i) * Sc;
  const int* er = eos + ((long)b * Sc + i) * Sc;
  float mx = -3.0e38f;
  for (int j = threadIdx.x; j < Sc; j += 256) {
    const bool m = (er[j] != 0) && (j == i - 1 || j == i + 1);
    mx = fmaxf(mx, m ? row[j] : NEGV);
  }
  mx = blk_reduce(mx, sh, true);
  float sum = 0.f;
  for (int j = threadIdx.x; j < Sc; j += 256) {
    const bool m = (er[j] != 0) && (j == i - 1 || j == i + 1);
    sum += expf((m ? row[j] : NEGV) - mx);
  }
  sum = blk_reduce(sum, sh, false);
  const float inv = 1.0f / sum;
  for (int j = threadIdx.x; j < Sc; j += 256) {
    const bool m = (er[j] != 0) && (j == i - 1 || j == i + 1);
    row[j] = expf((m ? row[j] : NEGV) - mx) * inv;
  }
}

// nb = prior + (1-prior)*sqrt(nb_sm * nb_sm^T + 1e-9)   -> d_out nb section
__global__ __launch_bounds__(256) void nb_kernel(
    const float* __restrict__ sm, const float* __restrict__ prior,
    float* __restrict__ nb)
{
  long idx = (long)blockIdx.x * 256 + threadIdx.x;
  if (idx >= (long)Bc * Sc * Sc) return;
  const int j = (int)(idx % Sc), i = (int)((idx / Sc) % Sc), b = (int)(idx / ((long)Sc * Sc));
  const float p = prior[0];
  const float v = sqrtf(sm[idx] * sm[((long)b * Sc + j) * Sc + i] + 1e-9f);
  nb[idx] = p + (1.0f - p) * v;
}

// L[b,i] = log(nb[b,i,i+1] + 1e-9)   (row S-1 contributes nothing)
__global__ __launch_bounds__(256) void l_kernel(
    const float* __restrict__ nb, float* __restrict__ L)
{
  int idx = blockIdx.x * 256 + threadIdx.x;
  if (idx >= Bc * Sc) return;
  const int i = idx % Sc, b = idx / Sc;
  L[idx] = (i < Sc - 1) ? logf(nb[((long)b * Sc + i) * Sc + (i + 1)] + 1e-9f) : 0.0f;
}

// exclusive prefix sum of L per batch (replaces the S^3 triangular matmuls)
__global__ __launch_bounds__(1024) void scan_kernel(
    const float* __restrict__ L, float* __restrict__ P)
{
  __shared__ float sh[Sc];
  const int b = blockIdx.x, t = threadIdx.x;
  sh[t] = L[b * Sc + t];
  __syncthreads();
  for (int off = 1; off < Sc; off <<= 1) {
    const float v = (t >= off) ? sh[t - off] : 0.0f;
    __syncthreads();
    sh[t] += v;
    __syncthreads();
  }
  P[b * Sc + t] = (t == 0) ? 0.0f : sh[t - 1];
}

// g_attn: off-diag exp(P[max]-P[min]) + 1e-9 ; diag nb[i,i]
__global__ __launch_bounds__(256) void gattn_kernel(
    const float* __restrict__ nb, const float* __restrict__ P,
    float* __restrict__ g)
{
  long idx = (long)blockIdx.x * 256 + threadIdx.x;
  if (idx >= (long)Bc * Sc * Sc) return;
  const int j = (int)(idx % Sc), i = (int)((idx / Sc) % Sc), b = (int)(idx / ((long)Sc * Sc));
  if (i == j) { g[idx] = nb[idx]; return; }
  const int lo = i < j ? i : j, hi = i < j ? j : i;
  g[idx] = expf(P[b * Sc + hi] - P[b * Sc + lo]) + 1e-9f;
}

// [B,S,D] f16 -> per-head layout. mode 0: [B,H,S,DK]; mode 1: [B,H,DK,S]
__global__ __launch_bounds__(256) void reshape_heads_kernel(
    const _Float16* __restrict__ src, _Float16* __restrict__ dst, int mode)
{
  long idx = (long)blockIdx.x * 256 + threadIdx.x;
  if (idx >= (long)Bc * Sc * Dc) return;
  const int dk = (int)(idx % DKc);
  const int h  = (int)((idx / DKc) % Hc);
  const int s  = (int)((idx / Dc) % Sc);
  const int b  = (int)(idx / ((long)Sc * Dc));
  const _Float16 v = src[idx];
  if (mode == 0) dst[(((long)b * Hc + h) * Sc + s) * DKc + dk] = v;
  else           dst[(((long)b * Hc + h) * DKc + dk) * Sc + s] = v;
}

// MHA row softmax (eos mask broadcast over heads) * g_attn -> f16 P matrix
__global__ __launch_bounds__(256) void mha_softmax_kernel(
    const float* __restrict__ sc, const int* __restrict__ eos,
    const float* __restrict__ gattn, _Float16* __restrict__ p16)
{
  __shared__ float sh[256];
  const int i = blockIdx.x, zh = blockIdx.y, b = zh / Hc;
  const float* row = sc + ((long)zh * Sc + i) * Sc;
  const int*   er  = eos + ((long)b * Sc + i) * Sc;
  const float* gr  = gattn + ((long)b * Sc + i) * Sc;
  float mx = -3.0e38f;
  for (int j = threadIdx.x; j < Sc; j += 256)
    mx = fmaxf(mx, (er[j] != 0) ? row[j] : NEGV);
  mx = blk_reduce(mx, sh, true);
  float sum = 0.f;
  for (int j = threadIdx.x; j < Sc; j += 256)
    sum += expf(((er[j] != 0) ? row[j] : NEGV) - mx);
  sum = blk_reduce(sum, sh, false);
  const float inv = 1.0f / sum;
  for (int j = threadIdx.x; j < Sc; j += 256) {
    const float v = expf(((er[j] != 0) ? row[j] : NEGV) - mx) * inv * gr[j];
    p16[((long)zh * Sc + i) * Sc + j] = (_Float16)v;
  }
}

// ------------------------------ host driver --------------------------------

static inline void launch_gemm(hipStream_t st, bool gelu,
    const _Float16* A, int lda, long sAb,
    const _Float16* Bt, int ldb, long sBb,
    const float* bias, const float* resid,
    float* Cf, _Float16* Ch, int ldc, int cDiv, long sCo, long sCi,
    int M, int N, int K, float alpha, int batches)
{
  dim3 grid((M / 128) * (N / 64), 1, batches);
  if (gelu)
    gemm_wmma<true><<<grid, 256, 0, st>>>(A, lda, sAb, Bt, ldb, sBb, bias, resid,
                                          Cf, Ch, ldc, cDiv, sCo, sCi, M, N, K, alpha);
  else
    gemm_wmma<false><<<grid, 256, 0, st>>>(A, lda, sAb, Bt, ldb, sBb, bias, resid,
                                           Cf, Ch, ldc, cDiv, sCo, sCi, M, N, K, alpha);
}

extern "C" void kernel_launch(void* const* d_in, const int* in_sizes, int n_in,
                              void* d_out, int out_size, void* d_ws, size_t ws_size,
                              hipStream_t stream)
{
  (void)in_sizes; (void)n_in; (void)out_size; (void)ws_size;
  const float* x      = (const float*)d_in[0];
  const int*   eos    = (const int*)  d_in[1];
  const float* prior  = (const float*)d_in[2];
  const float* ga_lna = (const float*)d_in[3];
  const float* ga_lnb = (const float*)d_in[4];
  const float* ga_wq  = (const float*)d_in[5];
  const float* ga_bq  = (const float*)d_in[6];
  const float* ga_wk  = (const float*)d_in[7];
  const float* ga_bk  = (const float*)d_in[8];
  const float* wq     = (const float*)d_in[9];
  const float* bq     = (const float*)d_in[10];
  const float* wk     = (const float*)d_in[11];
  const float* bk     = (const float*)d_in[12];
  const float* wv     = (const float*)d_in[13];
  const float* bv     = (const float*)d_in[14];
  const float* wo     = (const float*)d_in[15];
  const float* bo     = (const float*)d_in[16];
  const float* ln1a   = (const float*)d_in[17];
  const float* ln1b   = (const float*)d_in[18];
  const float* ln2a   = (const float*)d_in[19];
  const float* ln2b   = (const float*)d_in[20];
  const float* w1     = (const float*)d_in[21];
  const float* b1     = (const float*)d_in[22];
  const float* w2     = (const float*)d_in[23];
  const float* b2     = (const float*)d_in[24];

  float* out_x  = (float*)d_out;                                  // [B,S,D]
  float* out_g  = out_x + (long)Bc * Sc * Dc;                     // [B,S,S]
  float* out_nb = out_g + (long)Bc * Sc * Sc;                     // [B,S,S]

  // -------- workspace layout (MB offsets; total ~254 MB) --------
  char* ws = (char*)d_ws;
  const size_t MB = (size_t)1 << 20;
  _Float16* wt_gaq = (_Float16*)(ws + 0   * MB);  // [D,D]^T
  _Float16* wt_gak = (_Float16*)(ws + 2   * MB);
  _Float16* wt_q   = (_Float16*)(ws + 4   * MB);
  _Float16* wt_k   = (_Float16*)(ws + 6   * MB);
  _Float16* wt_v   = (_Float16*)(ws + 8   * MB);
  _Float16* wt_o   = (_Float16*)(ws + 10  * MB);
  _Float16* wt_1   = (_Float16*)(ws + 12  * MB);  // [DFF,D]
  _Float16* wt_2   = (_Float16*)(ws + 20  * MB);  // [D,DFF]
  _Float16* a16    = (_Float16*)(ws + 28  * MB);  // LN output f16 (reused)
  _Float16* q16    = (_Float16*)(ws + 36  * MB);  // [B,S,D]
  _Float16* k16    = (_Float16*)(ws + 44  * MB);
  _Float16* v16    = (_Float16*)(ws + 52  * MB);
  float*    sc_ga  = (float*)   (ws + 60  * MB);  // [B,S,S]
  _Float16* qh     = (_Float16*)(ws + 76  * MB);  // [B,H,S,DK]
  _Float16* kh     = (_Float16*)(ws + 84  * MB);
  _Float16* vt     = (_Float16*)(ws + 92  * MB);  // [B,H,DK,S]
  float*    sc2    = (float*)   (ws + 100 * MB);  // [B,H,S,S]
  _Float16* p16    = (_Float16*)(ws + 164 * MB);  // [B,H,S,S]
  _Float16* o16    = (_Float16*)(ws + 196 * MB);  // [B,S,D] head-merged
  float*    x1     = (float*)   (ws + 204 * MB);  // [B,S,D] post-attn residual
  _Float16* h16    = (_Float16*)(ws + 220 * MB);  // [B,S,DFF]
  float*    Lb     = (float*)   (ws + 252 * MB);  // [B,S]
  float*    Pb     = (float*)   (ws + 253 * MB);  // [B,S]

  const long BSD = (long)Bc * Sc * Dc, BSS = (long)Bc * Sc * Sc;
  const int BS = Bc * Sc;

  // ---- weight transpose+convert (f32 [K,N] -> f16 [N,K]) ----
  {
    const int nDD = (int)(((long)Dc * Dc + 255) / 256);
    tconv_kernel<<<nDD, 256, 0, stream>>>(ga_wq, wt_gaq, Dc, Dc);
    tconv_kernel<<<nDD, 256, 0, stream>>>(ga_wk, wt_gak, Dc, Dc);
    tconv_kernel<<<nDD, 256, 0, stream>>>(wq,    wt_q,   Dc, Dc);
    tconv_kernel<<<nDD, 256, 0, stream>>>(wk,    wt_k,   Dc, Dc);
    tconv_kernel<<<nDD, 256, 0, stream>>>(wv,    wt_v,   Dc, Dc);
    tconv_kernel<<<nDD, 256, 0, stream>>>(wo,    wt_o,   Dc, Dc);
    const int nDF = (int)(((long)Dc * DFFc + 255) / 256);
    tconv_kernel<<<nDF, 256, 0, stream>>>(w1, wt_1, Dc, DFFc);
    tconv_kernel<<<nDF, 256, 0, stream>>>(w2, wt_2, DFFc, Dc);
  }

  // ---- GroupAttention ----
  ln_kernel<<<BS, 256, 0, stream>>>(x, ga_lna, ga_lnb, a16, Dc);
  launch_gemm(stream, false, a16, Dc, 0, wt_gaq, Dc, 0, ga_bq, nullptr,
              nullptr, q16, Dc, 1, 0, 0, BS, Dc, Dc, 1.0f, 1);
  launch_gemm(stream, false, a16, Dc, 0, wt_gak, Dc, 0, ga_bk, nullptr,
              nullptr, k16, Dc, 1, 0, 0, BS, Dc, Dc, 1.0f, 1);
  // scores[b] = q[b] @ k[b]^T / 256
  launch_gemm(stream, false, q16, Dc, (long)Sc * Dc, k16, Dc, (long)Sc * Dc,
              nullptr, nullptr, sc_ga, nullptr, Sc, 1, (long)Sc * Sc, 0,
              Sc, Sc, Dc, 1.0f / 256.0f, Bc);
  ga_softmax_kernel<<<dim3(Sc, Bc), 256, 0, stream>>>(sc_ga, eos);
  nb_kernel<<<(int)((BSS + 255) / 256), 256, 0, stream>>>(sc_ga, prior, out_nb);
  l_kernel<<<(BS + 255) / 256, 256, 0, stream>>>(out_nb, Lb);
  scan_kernel<<<Bc, Sc, 0, stream>>>(Lb, Pb);
  gattn_kernel<<<(int)((BSS + 255) / 256), 256, 0, stream>>>(out_nb, Pb, out_g);

  // ---- Self-attention sublayer ----
  ln_kernel<<<BS, 256, 0, stream>>>(x, ln1a, ln1b, a16, Dc);
  launch_gemm(stream, false, a16, Dc, 0, wt_q, Dc, 0, bq, nullptr,
              nullptr, q16, Dc, 1, 0, 0, BS, Dc, Dc, 1.0f, 1);
  launch_gemm(stream, false, a16, Dc, 0, wt_k, Dc, 0, bk, nullptr,
              nullptr, k16, Dc, 1, 0, 0, BS, Dc, Dc, 1.0f, 1);
  launch_gemm(stream, false, a16, Dc, 0, wt_v, Dc, 0, bv, nullptr,
              nullptr, v16, Dc, 1, 0, 0, BS, Dc, Dc, 1.0f, 1);
  {
    const int nBSD = (int)((BSD + 255) / 256);
    reshape_heads_kernel<<<nBSD, 256, 0, stream>>>(q16, qh, 0);
    reshape_heads_kernel<<<nBSD, 256, 0, stream>>>(k16, kh, 0);
    reshape_heads_kernel<<<nBSD, 256, 0, stream>>>(v16, vt, 1);
  }
  // sc2[b,h] = qh @ kh^T / sqrt(DK)
  launch_gemm(stream, false, qh, DKc, (long)Sc * DKc, kh, DKc, (long)Sc * DKc,
              nullptr, nullptr, sc2, nullptr, Sc, 1, (long)Sc * Sc, 0,
              Sc, Sc, DKc, 0.125f, Bc * Hc);
  mha_softmax_kernel<<<dim3(Sc, Bc * Hc), 256, 0, stream>>>(sc2, eos, out_g, p16);
  // o[b,h] = p @ v  -> stored head-merged into o16 [B*S, D]
  launch_gemm(stream, false, p16, Sc, (long)Sc * Sc, vt, Sc, (long)DKc * Sc,
              nullptr, nullptr, nullptr, o16, Dc, Hc, (long)Sc * Dc, DKc,
              Sc, DKc, Sc, 1.0f, Bc * Hc);
  // x1 = x + o @ wo + bo
  launch_gemm(stream, false, o16, Dc, 0, wt_o, Dc, 0, bo, x,
              x1, nullptr, Dc, 1, 0, 0, BS, Dc, Dc, 1.0f, 1);

  // ---- FFN sublayer ----
  ln_kernel<<<BS, 256, 0, stream>>>(x1, ln2a, ln2b, a16, Dc);
  launch_gemm(stream, true, a16, Dc, 0, wt_1, Dc, 0, b1, nullptr,
              nullptr, h16, DFFc, 1, 0, 0, BS, DFFc, Dc, 1.0f, 1);
  launch_gemm(stream, false, h16, DFFc, 0, wt_2, DFFc, 0, b2, x1,
              out_x, nullptr, Dc, 1, 0, 0, BS, Dc, DFFc, 1.0f, 1);
}
